// MHA_79654463471785
// MI455X (gfx1250) — matrix operand order repaired and checked by reading, not compiled, not run on hardware
//
#include <hip/hip_runtime.h>
#include <hip/hip_bf16.h>

// ---------------------------------------------------------------------------
// Causal MHA forward for MI455X (gfx1250), f16 WMMA with f32 accumulation.
// Pipeline: cvt(x,Wqkv,Wp) -> QKV GEMM -> flash attention -> proj GEMM.
// Round 2: straight-copy tiles now use the CDNA5 async global->LDS path
// (global_load_async_to_lds_b128, ASYNCcnt, s_wait_asynccnt).
// ---------------------------------------------------------------------------

typedef __attribute__((ext_vector_type(16))) _Float16 v16h;
typedef __attribute__((ext_vector_type(8)))  float    v8f;

union V8f  { v8f v; float e[8]; };
union H4   { uint2 u; _Float16 h[4]; };
union F16x16 { v16h v; uint4 q[2]; };

static constexpr int S_LEN = 4096;
static constexpr int EMB   = 1024;
static constexpr int NHEAD = 16;
static constexpr int DHEAD = 64;

// ---- CDNA5 async global->LDS copy (16B per lane, tracked by ASYNCcnt) -----
// Flat shared addresses have the LDS byte offset in their low 32 bits
// (ISA 10.2 aperture rules), which is what the
// async instruction's VDST operand wants.
__device__ __forceinline__ void async_copy_b128(void* lds_dst, const void* g_src) {
    unsigned lds = (unsigned)(size_t)lds_dst;
    asm volatile("global_load_async_to_lds_b128 %0, %1, off"
                 :: "v"(lds), "v"(g_src)
                 : "memory");
}
__device__ __forceinline__ void wait_async0() {
    asm volatile("s_wait_asynccnt 0" ::: "memory");
}

// ---- WMMA fragment loaders (wave32 layouts per CDNA5 ISA 7.12.2) ----------
// A (16x32 f16): lane L -> row = L%16; elems 0..7 = K (hi..hi+7),
//                elems 8..15 = K (16+hi..23+hi), hi = (L<16)?0:8.
__device__ __forceinline__ v16h ld_frag_a(const _Float16* rowp, int hi) {
    F16x16 u;
    u.q[0] = *(const uint4*)(rowp + hi);
    u.q[1] = *(const uint4*)(rowp + 16 + hi);
    return u.v;
}
// B (32x16 f16): lane L -> col = L%16; elems 0..7 = K (hi..hi+7),
//                elems 8..15 = K (16+hi..23+hi), hi = kb/2, kb = (L<16)?0:16.  Requires LDS stored as [n][k].
__device__ __forceinline__ v16h ld_frag_b(const _Float16* colp, int kb) {
    F16x16 u;
    u.q[0] = *(const uint4*)(colp + (kb >> 1));
    u.q[1] = *(const uint4*)(colp + 16 + (kb >> 1));
    return u.v;
}

__device__ __forceinline__ v8f wmma_f16(v16h a, v16h b, v8f c) {
    return __builtin_amdgcn_wmma_f32_16x16x32_f16(
        /*neg_a=*/false, a, /*neg_b=*/false, b,
        /*c_mod=*/(short)0, c, /*reuse_a=*/false, /*reuse_b=*/false);
}

// ---------------------------------------------------------------------------
// Kernel 1: fp32 -> f16 conversion (vectorized by 4; all sizes are mult of 4)
// ---------------------------------------------------------------------------
__global__ void cvt_f32_to_f16(const float* __restrict__ in,
                               _Float16* __restrict__ out, long n) {
    long i = ((long)blockIdx.x * blockDim.x + threadIdx.x) * 4;
    long stride = (long)gridDim.x * blockDim.x * 4;
    for (; i + 3 < n; i += stride) {
        float4 v = *(const float4*)(in + i);
        H4 t;
        t.h[0] = (_Float16)v.x; t.h[1] = (_Float16)v.y;
        t.h[2] = (_Float16)v.z; t.h[3] = (_Float16)v.w;
        *(uint2*)(out + i) = t.u;
    }
}

// ---------------------------------------------------------------------------
// Kernel 2/4: f16 GEMM  C[M,N] = A[M,K] @ B[K,N] + bias
//   Block tile 128x64, BK=32, 256 threads = 8 waves in a 4(M) x 2(N) grid,
//   each wave computes a 32x32 patch = 2x2 WMMA 16x16 tiles.
//   A tile copied via async global->LDS; B tile transposed via registers.
//   MODE 0: scatter f16 into qkv buffer [3][H][S][D]   (QKV projection)
//   MODE 1: store f32 row-major                         (output projection)
// ---------------------------------------------------------------------------
template<int MODE>
__global__ __launch_bounds__(256) void gemm_f16_wmma(
        const _Float16* __restrict__ A, const _Float16* __restrict__ B,
        const float* __restrict__ bias, void* __restrict__ Cout,
        int M, int N, int K) {
    constexpr int BM = 128, BN = 64, LDS_S = 40;   // pad 32->40 halfs
    __shared__ __align__(16) _Float16 Alds[BM][LDS_S];
    __shared__ __align__(16) _Float16 Blds[BN][LDS_S];   // transposed: [n][k]

    const int tid   = threadIdx.x;
    const int wave  = tid >> 5;
    const int lane  = tid & 31;
    const int l16   = lane & 15;
    const int hiA   = (lane < 16) ? 0 : 8;
    const int kbB   = (lane < 16) ? 0 : 16;
    const int hi8   = (lane < 16) ? 0 : 8;
    const int mW    = (wave >> 1) * 32;
    const int nW    = (wave & 1) * 32;
    const int m0    = blockIdx.x * BM;
    const int n0    = blockIdx.y * BN;

    v8f zero = {};
    V8f acc[2][2];
#pragma unroll
    for (int a = 0; a < 2; ++a)
#pragma unroll
        for (int b = 0; b < 2; ++b) acc[a][b].v = zero;

    for (int k0 = 0; k0 < K; k0 += 32) {
        __syncthreads();
        // A tile: 128x32 halfs = 512 x 16B chunks; async global->LDS copy.
#pragma unroll
        for (int i = 0; i < 2; ++i) {
            int ch = tid * 2 + i;              // 0..511
            int r = ch >> 2, c8 = (ch & 3) * 8;
            async_copy_b128(&Alds[r][c8],
                            &A[(size_t)(m0 + r) * K + k0 + c8]);
        }
        // B tile: 32x64 loaded coalesced, stored transposed into [n][k].
#pragma unroll
        for (int i = 0; i < 2; ++i) {
            int ch = tid * 2 + i;              // 0..511
            int kk = ch >> 4, n4 = (ch & 15) * 4;
            H4 t;
            t.u = *(const uint2*)&B[(size_t)(k0 + kk) * N + n0 + n4];
#pragma unroll
            for (int j = 0; j < 4; ++j) Blds[n4 + j][kk] = t.h[j];
        }
        if (k0 + 32 < K) {   // gfx1250 global_prefetch_b8 path
            __builtin_prefetch(&A[(size_t)(m0 + (tid >> 1)) * K + k0 + 32], 0, 0);
            __builtin_prefetch(&B[(size_t)(k0 + 32 + (tid & 31)) * N + n0], 0, 0);
        }
        wait_async0();                         // our async writes done
        __syncthreads();                       // everyone's writes visible

        v16h aF[2], bF[2];
        aF[0] = ld_frag_a(&Alds[mW + l16][0], hiA);
        aF[1] = ld_frag_a(&Alds[mW + 16 + l16][0], hiA);
        bF[0] = ld_frag_b(&Blds[nW + l16][0], kbB);
        bF[1] = ld_frag_b(&Blds[nW + 16 + l16][0], kbB);
#pragma unroll
        for (int mw = 0; mw < 2; ++mw)
#pragma unroll
            for (int nw = 0; nw < 2; ++nw)
                acc[mw][nw].v = wmma_f16(aF[mw], bF[nw], acc[mw][nw].v);
    }

    // Epilogue. C/D layout: elem r -> row r (+8 for hi lanes), col = lane%16.
#pragma unroll
    for (int mw = 0; mw < 2; ++mw) {
#pragma unroll
        for (int nw = 0; nw < 2; ++nw) {
            int col = n0 + nW + nw * 16 + l16;
            float bv = bias[col];
#pragma unroll
            for (int r = 0; r < 8; ++r) {
                int row = m0 + mW + mw * 16 + r + hi8;
                float val = acc[mw][nw].e[r] + bv;
                if (MODE == 0) {
                    int part = col >> 10;         // 0:Q 1:K 2:V
                    int rem  = col & 1023;
                    int hh   = rem >> 6;
                    int dd   = rem & 63;
                    ((_Float16*)Cout)[(((size_t)part * NHEAD + hh) * S_LEN + row)
                                      * DHEAD + dd] = (_Float16)val;
                } else {
                    ((float*)Cout)[(size_t)row * N + col] = val;
                }
            }
        }
    }
}

// ---------------------------------------------------------------------------
// Kernel 3: flash attention (causal, no 1/sqrt(D) scaling — faithful).
//   grid = (S/128, H); block = 256 (8 waves). Each wave owns 16 q-rows, so
//   softmax row statistics reduce within a 16-lane half via shfl_xor.
//   Q and K tiles use async global->LDS; V needs a transpose so it goes
//   through registers.
// ---------------------------------------------------------------------------
__global__ __launch_bounds__(256) void flash_attn_f16(
        const _Float16* __restrict__ qkv,   // [3][H][S][D] f16
        _Float16* __restrict__ o16) {       // [S][E] f16
    __shared__ __align__(16) _Float16 Qlds[128][72];
    __shared__ __align__(16) _Float16 Klds[64][72];      // [kv][d]
    __shared__ __align__(16) _Float16 Vlds[64][72];      // [d][kv] (transposed)
    __shared__ __align__(16) _Float16 Plds[8][16][72];   // per-wave P patch

    const int tid  = threadIdx.x;
    const int w    = tid >> 5;
    const int lane = tid & 31;
    const int l16  = lane & 15;
    const int hiA  = (lane < 16) ? 0 : 8;
    const int kbB  = (lane < 16) ? 0 : 16;
    const int hi8  = (lane < 16) ? 0 : 8;
    const int qb   = blockIdx.x;
    const int h    = blockIdx.y;

    const _Float16* Qp = qkv + ((size_t)0 * NHEAD + h) * S_LEN * DHEAD;
    const _Float16* Kp = qkv + ((size_t)1 * NHEAD + h) * S_LEN * DHEAD;
    const _Float16* Vp = qkv + ((size_t)2 * NHEAD + h) * S_LEN * DHEAD;

    // Load Q tile: 128 rows x 64 halfs = 1024 x 16B chunks, async copies.
#pragma unroll
    for (int i = 0; i < 4; ++i) {
        int ch = tid * 4 + i;                 // 0..1023
        int r = ch >> 3, d8 = (ch & 7) * 8;
        async_copy_b128(&Qlds[r][d8],
                        &Qp[(size_t)(qb * 128 + r) * DHEAD + d8]);
    }

    float m_r[8], l_r[8];
    V8f   Oacc[4];
    v8f   zero = {};
#pragma unroll
    for (int r = 0; r < 8; ++r) { m_r[r] = -3.0e38f; l_r[r] = 0.0f; }
#pragma unroll
    for (int n = 0; n < 4; ++n) Oacc[n].v = zero;

    wait_async0();                            // Q tile resident in LDS

    const int c_end = qb * 128 + 128;         // exclusive; causal bound
    for (int c0 = 0; c0 < c_end; c0 += 64) {
        __syncthreads();
        // K tile (row-major [kv][d]): 512 x 16B chunks, async copies.
#pragma unroll
        for (int i = 0; i < 2; ++i) {
            int ch = tid * 2 + i;             // 0..511
            int c = ch >> 3, d8 = (ch & 7) * 8;
            async_copy_b128(&Klds[c][d8],
                            &Kp[(size_t)(c0 + c) * DHEAD + d8]);
        }
        // V tile transposed into [d][kv]: 1024 x 8B chunks via registers.
#pragma unroll
        for (int i = 0; i < 4; ++i) {
            int ch = tid * 4 + i;             // 0..1023
            int c = ch >> 4, d4 = (ch & 15) * 4;
            H4 t;
            t.u = *(const uint2*)&Vp[(size_t)(c0 + c) * DHEAD + d4];
#pragma unroll
            for (int j = 0; j < 4; ++j) Vlds[d4 + j][c] = t.h[j];
        }
        wait_async0();                        // our K async writes done
        __syncthreads();                      // all K/V writes visible

        // ---- S = Q @ K^T  (16 rows x 64 cols per wave) ----
        V8f sacc[4];
#pragma unroll
        for (int n = 0; n < 4; ++n) sacc[n].v = zero;
        v16h aQ[2];
        aQ[0] = ld_frag_a(&Qlds[w * 16 + l16][0],  hiA);
        aQ[1] = ld_frag_a(&Qlds[w * 16 + l16][32], hiA);
#pragma unroll
        for (int n = 0; n < 4; ++n) {
            v16h b0 = ld_frag_b(&Klds[n * 16 + l16][0],  kbB);
            v16h b1 = ld_frag_b(&Klds[n * 16 + l16][32], kbB);
            sacc[n].v = wmma_f16(aQ[0], b0, sacc[n].v);
            sacc[n].v = wmma_f16(aQ[1], b1, sacc[n].v);
        }

        // ---- causal mask (wave-uniform gate, elementwise select) ----
        const int qbase = qb * 128 + w * 16 + hi8;     // + r = this row
        if (c0 + 63 > qb * 128 + w * 16) {
#pragma unroll
            for (int n = 0; n < 4; ++n) {
                int col = c0 + n * 16 + l16;
#pragma unroll
                for (int r = 0; r < 8; ++r)
                    sacc[n].e[r] = (col > qbase + r) ? -1.0e30f : sacc[n].e[r];
            }
        }

        // ---- online softmax: row max over 64 cols (4 tiles x 16 lanes) ----
        float tmax[8];
#pragma unroll
        for (int r = 0; r < 8; ++r) {
            float t = sacc[0].e[r];
            t = fmaxf(t, sacc[1].e[r]);
            t = fmaxf(t, sacc[2].e[r]);
            t = fmaxf(t, sacc[3].e[r]);
            tmax[r] = t;
        }
#pragma unroll
        for (int msk = 1; msk <= 8; msk <<= 1)
#pragma unroll
            for (int r = 0; r < 8; ++r)
                tmax[r] = fmaxf(tmax[r], __shfl_xor(tmax[r], msk, 32));

        float scale[8];
#pragma unroll
        for (int r = 0; r < 8; ++r) {
            float mn = fmaxf(m_r[r], tmax[r]);
            scale[r] = __expf(m_r[r] - mn);
            m_r[r] = mn;
        }

        // ---- P = exp(S - m), row sums, stage P for the PV WMMA ----
        float rsum[8];
#pragma unroll
        for (int r = 0; r < 8; ++r) rsum[r] = 0.0f;
#pragma unroll
        for (int n = 0; n < 4; ++n) {
#pragma unroll
            for (int r = 0; r < 8; ++r) {
                float pv = __expf(sacc[n].e[r] - m_r[r]);
                rsum[r] += pv;
                Plds[w][r + hi8][n * 16 + l16] = (_Float16)pv;
            }
        }
#pragma unroll
        for (int msk = 1; msk <= 8; msk <<= 1)
#pragma unroll
            for (int r = 0; r < 8; ++r)
                rsum[r] += __shfl_xor(rsum[r], msk, 32);
#pragma unroll
        for (int r = 0; r < 8; ++r) l_r[r] = l_r[r] * scale[r] + rsum[r];
#pragma unroll
        for (int n = 0; n < 4; ++n)
#pragma unroll
            for (int r = 0; r < 8; ++r) Oacc[n].e[r] *= scale[r];

        // ---- O += P @ V (same-wave LDS producer/consumer; DS in-order) ----
#pragma unroll
        for (int kki = 0; kki < 2; ++kki) {
            v16h aP = ld_frag_a(&Plds[w][l16][kki * 32], hiA);
#pragma unroll
            for (int n = 0; n < 4; ++n) {
                v16h bV = ld_frag_b(&Vlds[n * 16 + l16][kki * 32], kbB);
                Oacc[n].v = wmma_f16(aP, bV, Oacc[n].v);
            }
        }
    }

    // ---- normalize and store O (f16, row-major [S][E]) ----
    float rinv[8];
#pragma unroll
    for (int r = 0; r < 8; ++r) rinv[r] = 1.0f / l_r[r];
#pragma unroll
    for (int n = 0; n < 4; ++n) {
        int e = h * DHEAD + n * 16 + l16;
#pragma unroll
        for (int r = 0; r < 8; ++r) {
            int row = qb * 128 + w * 16 + r + hi8;
            o16[(size_t)row * EMB + e] = (_Float16)(Oacc[n].e[r] * rinv[r]);
        }
    }
}

// ---------------------------------------------------------------------------
// Host-side orchestration.
// Workspace layout (bytes), total ~48 MB:
//   [0          ) x16    : S*E*2       =  8,388,608
//   [ 8,388,608 ) wqkv16 : E*3E*2      =  6,291,456
//   [14,680,064 ) wp16   : E*E*2       =  2,097,152
//   [16,777,216 ) qkv16  : 3*S*E*2     = 25,165,824   [3][H][S][D]
//   [41,943,040 ) o16    : S*E*2       =  8,388,608
// ---------------------------------------------------------------------------
extern "C" void kernel_launch(void* const* d_in, const int* in_sizes, int n_in,
                              void* d_out, int out_size, void* d_ws, size_t ws_size,
                              hipStream_t stream) {
    const float* x    = (const float*)d_in[0];
    /* d_in[1] = causal mask (bool) — recomputed analytically, unused */
    const float* Wqkv = (const float*)d_in[2];
    const float* bqkv = (const float*)d_in[3];
    const float* Wp   = (const float*)d_in[4];
    const float* bp   = (const float*)d_in[5];

    char* ws = (char*)d_ws;
    _Float16* x16    = (_Float16*)(ws);
    _Float16* wqkv16 = (_Float16*)(ws +  8388608);
    _Float16* wp16   = (_Float16*)(ws + 14680064);
    _Float16* qkv16  = (_Float16*)(ws + 16777216);
    _Float16* o16    = (_Float16*)(ws + 41943040);

    auto cvt_blocks = [](long n) {
        long b = (n / 4 + 255) / 256;
        return (int)(b > 4096 ? 4096 : b);
    };
    long nx = (long)S_LEN * EMB, nwq = (long)EMB * 3 * EMB, nwp = (long)EMB * EMB;
    cvt_f32_to_f16<<<cvt_blocks(nx),  256, 0, stream>>>(x,    x16,    nx);
    cvt_f32_to_f16<<<cvt_blocks(nwq), 256, 0, stream>>>(Wqkv, wqkv16, nwq);
    cvt_f32_to_f16<<<cvt_blocks(nwp), 256, 0, stream>>>(Wp,   wp16,   nwp);

    // QKV projection: [4096,1024] @ [1024,3072]
    gemm_f16_wmma<0><<<dim3(S_LEN / 128, (3 * EMB) / 64), 256, 0, stream>>>(
        x16, wqkv16, bqkv, (void*)qkv16, S_LEN, 3 * EMB, EMB);

    // Flash attention: grid (S/128 q-blocks, H heads)
    flash_attn_f16<<<dim3(S_LEN / 128, NHEAD), 256, 0, stream>>>(qkv16, o16);

    // Output projection: [4096,1024] @ [1024,1024] -> fp32 d_out
    gemm_f16_wmma<1><<<dim3(S_LEN / 128, EMB / 64), 256, 0, stream>>>(
        o16, wp16, bp, d_out, S_LEN, EMB, EMB);
}
